// RNNAttention_75239237091381
// MI455X (gfx1250) — matrix operand order repaired and checked by reading, not compiled, not run on hardware
//
#include <hip/hip_runtime.h>
#include <hip/hip_bf16.h>

typedef __attribute__((ext_vector_type(16))) __bf16 v16bf;
typedef __attribute__((ext_vector_type(8)))  __bf16 v8bf;
typedef __attribute__((ext_vector_type(8)))  float  v8f;
typedef unsigned int u32;
typedef __attribute__((ext_vector_type(4))) unsigned int u32x4;
typedef __attribute__((ext_vector_type(8))) int i32x8;
typedef __attribute__((ext_vector_type(4))) int i32x4;

#define B_ 64
#define T_ 512
#define V_ 512
#define E_ 128
#define D_ 256

#if defined(__has_builtin)
#if __has_builtin(__builtin_amdgcn_tensor_load_to_lds)
#define HAVE_TDM 1
#endif
#endif
#if defined(HAVE_TDM) && defined(__has_include)
#if __has_include(<hip/amd_detail/amd_gfx1250_TDM.h>)
#define TDM_ARGS6 1
#endif
#endif

static __device__ inline v8f wmma_bf16(v16bf a, v16bf b, v8f c) {
  return __builtin_amdgcn_wmma_f32_16x16x32_bf16(false, a, false, b, (short)0, c,
                                                 false, false);
}

// A fragment: row-major A, 16x32 bf16 tile at (m0,k0).
// Lanes 0-15: row m0+lane, K = k0+0..7 and k0+16..23
// Lanes 16-31: row m0+lane-16, K = k0+8..15 and k0+24..31
static __device__ inline v16bf load_a16(const __bf16* A, int lda, int m0, int k0,
                                        int lane) {
  const __bf16* p = A + (size_t)(m0 + (lane & 15)) * lda + k0 + ((lane >> 4) << 3);
  v8bf lo = *(const v8bf*)p;
  v8bf hi = *(const v8bf*)(p + 16);
  return __builtin_shufflevector(lo, hi, 0, 1, 2, 3, 4, 5, 6, 7, 8, 9, 10, 11, 12,
                                 13, 14, 15);
}

// B fragment from TRANSPOSED weights WT[N][K]: 32x16 bf16 tile at (k0,n0).
// Lane L: column n0+(L&15), K = k0 + (L>=16 ? 8 : 0) + 0..7 and +16..23
static __device__ inline v16bf load_bT16(const __bf16* WT, int ldk, int n0, int k0,
                                         int lane) {
  const __bf16* p = WT + (size_t)(n0 + (lane & 15)) * ldk + k0 + ((lane >> 4) << 3);
  v8bf lo = *(const v8bf*)p;
  v8bf hi = *(const v8bf*)(p + 16);
  return __builtin_shufflevector(lo, hi, 0, 1, 2, 3, 4, 5, 6, 7, 8, 9, 10, 11, 12,
                                 13, 14, 15);
}

// TDM: DMA a [rows x row_elems] bf16 tile (contiguous, stride=row_elems) into
// LDS. Issued by one wave; caller must __syncthreads() after (issuing wave
// waits TENSORcnt==0 before the barrier). D# layout per CDNA5 ISA.
static __device__ inline void tdm_load_panel(const __bf16* gsrc,
                                             __bf16* lds_dst, int rows,
                                             int row_elems) {
#ifdef HAVE_TDM
  unsigned lds_off = (unsigned)(unsigned long long)(size_t)lds_dst;
  unsigned long long ga = (unsigned long long)(size_t)gsrc;
  u32x4 g0;
  g0[0] = 1u;                                   // count=1, load, user mode
  g0[1] = lds_off;                              // lds_addr
  g0[2] = (u32)(ga & 0xFFFFFFFFu);              // global_addr[31:0]
  g0[3] = (u32)((ga >> 32) & 0x01FFFFFFu) | (2u << 30);  // addr[56:32], type=2
  i32x8 g1;
  g1[0] = (int)(1u << 16);                      // wg_mask=0, data_size=1 (2B)
  g1[1] = (int)((u32)row_elems << 16);          // tensor_dim0[15:0]
  g1[2] = (int)((u32)rows << 16);               // tensor_dim0[31:16]=0, dim1 lo
  g1[3] = (int)((u32)row_elems << 16);          // dim1 hi=0, tile_dim0
  g1[4] = rows;                                 // tile_dim1=rows, tile_dim2=0
  g1[5] = row_elems;                            // tensor_dim0_stride[31:0]
  g1[6] = 0;
  g1[7] = 0;
  i32x4 z4 = {0, 0, 0, 0};
#ifdef TDM_ARGS6
  i32x8 z8 = {0, 0, 0, 0, 0, 0, 0, 0};
  __builtin_amdgcn_tensor_load_to_lds(g0, g1, z4, z4, z8, 0);
#else
  __builtin_amdgcn_tensor_load_to_lds(g0, g1, z4, z4, 0);
#endif
  __builtin_amdgcn_s_wait_tensorcnt((short)0);
#else
  (void)gsrc; (void)lds_dst; (void)rows; (void)row_elems;
#endif
}

__global__ void k_cvt_bf16(const float* __restrict__ in, __bf16* __restrict__ out,
                           int n) {
  int i = blockIdx.x * blockDim.x + threadIdx.x;
  if (i < n) out[i] = (__bf16)in[i];
}

// in: fp32 [K x N] row-major  ->  out: bf16 [N x K] row-major
__global__ void k_transpose_bf16(const float* __restrict__ in,
                                 __bf16* __restrict__ out, int K, int N) {
  int i = blockIdx.x * blockDim.x + threadIdx.x;
  if (i < K * N) {
    int n = i / K, k = i % K;
    out[i] = (__bf16)in[k * N + n];
  }
}

// XP[t*B+b][d] = embed[x[b*T+t]] @ W_ih + b_h ; M-row index m = b*T+t
__global__ void __launch_bounds__(256) k_embed_proj(
    const int* __restrict__ x, const __bf16* __restrict__ Ebf,
    const __bf16* __restrict__ WihT, const float* __restrict__ b_h,
    float* __restrict__ XP) {
  int lane = threadIdx.x & 31, wave = threadIdx.x >> 5;
  int m0 = blockIdx.x * 16;  // 2048 blocks cover M = 32768
  for (int nt = wave * 2; nt < wave * 2 + 2; ++nt) {  // 16 N-tiles
    v8f acc = {};
#pragma unroll
    for (int k0 = 0; k0 < E_; k0 += 32) {
      int row = m0 + (lane & 15);
      int tok = x[row];
      const __bf16* p = Ebf + (size_t)tok * E_ + k0 + ((lane >> 4) << 3);
      v8bf lo = *(const v8bf*)p;
      v8bf hi = *(const v8bf*)(p + 16);
      v16bf a = __builtin_shufflevector(lo, hi, 0, 1, 2, 3, 4, 5, 6, 7, 8, 9, 10,
                                        11, 12, 13, 14, 15);
      v16bf b = load_bT16(WihT, E_, nt * 16, k0, lane);
      acc = wmma_bf16(a, b, acc);
    }
    int n = nt * 16 + (lane & 15);
    int mB = m0 + ((lane >> 4) << 3);
    float bias = b_h[n];
#pragma unroll
    for (int v = 0; v < 8; ++v) {
      int m = mB + v;
      int bb = m >> 9, tt = m & 511;  // m = b*T + t
      XP[((size_t)tt * B_ + bb) * D_ + n] = acc[v] + bias;
    }
  }
}

// One workgroup: h=tanh(xp + h_prev@W_hh); then proj=h@U_attn, P_t=h@W_attn.
__global__ void __launch_bounds__(256) k_step(
    int t, const float* __restrict__ XP, const __bf16* __restrict__ Hprev,
    const __bf16* __restrict__ WhhT, const __bf16* __restrict__ UattnT,
    const __bf16* __restrict__ WattnT, const float* __restrict__ b_h,
    __bf16* __restrict__ Hbf_t, float* __restrict__ P_t, float* __restrict__ Proj,
    __bf16* __restrict__ HC_t) {
  __shared__ __bf16 h_s[B_ * D_];   // 32 KB: this step's h (A operand, GEMM 2/3)
  __shared__ __bf16 hp_s[B_ * D_];  // 32 KB: h_prev panel (A operand, GEMM 1)
  int lane = threadIdx.x & 31, wave = threadIdx.x >> 5;
  if (t > 0) {
#ifdef HAVE_TDM
    if (wave == 0) tdm_load_panel(Hprev, hp_s, B_, D_);
#else
    for (int i = threadIdx.x * 8; i < B_ * D_; i += 256 * 8)
      *(v8bf*)(hp_s + i) = *(const v8bf*)(Hprev + i);
#endif
    __syncthreads();
  }
  for (int nt = wave * 2; nt < wave * 2 + 2; ++nt) {
    for (int mt = 0; mt < 4; ++mt) {
      v8f acc = {};
      if (t > 0) {
#pragma unroll
        for (int k0 = 0; k0 < D_; k0 += 32) {
          v16bf a = load_a16(hp_s, D_, mt * 16, k0, lane);
          v16bf b = load_bT16(WhhT, D_, nt * 16, k0, lane);
          acc = wmma_bf16(a, b, acc);
        }
      }
      int n = nt * 16 + (lane & 15);
      int mB = mt * 16 + ((lane >> 4) << 3);
#pragma unroll
      for (int v = 0; v < 8; ++v) {
        int m = mB + v;
        float h = tanhf(XP[((size_t)t * B_ + m) * D_ + n] + b_h[n] + acc[v]);
        __bf16 hb = (__bf16)h;
        h_s[m * D_ + n] = hb;
        Hbf_t[m * D_ + n] = hb;
        HC_t[m * 2 * D_ + n] = hb;  // first half of concat(h, ctx)
      }
    }
  }
  __syncthreads();
  for (int g = 0; g < 2; ++g) {
    const __bf16* WT = g ? WattnT : UattnT;
    float* Out = g ? P_t : Proj;
    for (int nt = wave * 2; nt < wave * 2 + 2; ++nt) {
      for (int mt = 0; mt < 4; ++mt) {
        v8f acc = {};
#pragma unroll
        for (int k0 = 0; k0 < D_; k0 += 32) {
          v16bf a = load_a16(h_s, D_, mt * 16, k0, lane);
          v16bf b = load_bT16(WT, D_, nt * 16, k0, lane);
          acc = wmma_bf16(a, b, acc);
        }
        int n = nt * 16 + (lane & 15);
        int mB = mt * 16 + ((lane >> 4) << 3);
#pragma unroll
        for (int v = 0; v < 8; ++v) Out[(mB + v) * D_ + n] = acc[v];
      }
    }
  }
}

// Per batch element: energy over past steps, softmax, context vector.
__global__ void __launch_bounds__(256) k_attn(int t, const float* __restrict__ P,
                                              const float* __restrict__ Proj,
                                              const __bf16* __restrict__ Hbf,
                                              const float* __restrict__ v_attn,
                                              __bf16* __restrict__ HC) {
  int b = blockIdx.x;
  int tid = threadIdx.x;
  __bf16* ctx_out = HC + ((size_t)t * B_ + b) * (2 * D_) + D_;
  if (t == 0) {  // ctx = h
    ctx_out[tid] = Hbf[(size_t)b * D_ + tid];
    return;
  }
  __shared__ float energy[T_];
  __shared__ float proj_s[D_];
  __shared__ float vs[D_];
  __shared__ float red[256];
  proj_s[tid] = Proj[b * D_ + tid];
  vs[tid] = v_attn[tid];
  __syncthreads();
  int lane = tid & 31, wave = tid >> 5;
  for (int tp = wave; tp < t; tp += 8) {
    if (lane == 0 && tp + 8 < t)  // pull next chunk toward L0 ahead of use
      __builtin_prefetch(P + ((size_t)(tp + 8) * B_ + b) * D_, 0, 1);
    const float* Prow = P + ((size_t)tp * B_ + b) * D_ + lane * 8;
    float part = 0.f;
#pragma unroll
    for (int j = 0; j < 8; ++j)
      part += vs[lane * 8 + j] * tanhf(Prow[j] + proj_s[lane * 8 + j]);
#pragma unroll
    for (int off = 16; off > 0; off >>= 1) part += __shfl_xor(part, off, 32);
    if (lane == 0) energy[tp] = part;
  }
  __syncthreads();
  float m = -3.0e38f;
  for (int i = tid; i < t; i += 256) m = fmaxf(m, energy[i]);
  red[tid] = m;
  __syncthreads();
  for (int s = 128; s > 0; s >>= 1) {
    if (tid < s) red[tid] = fmaxf(red[tid], red[tid + s]);
    __syncthreads();
  }
  float maxv = red[0];
  __syncthreads();
  float sum = 0.f;
  for (int i = tid; i < t; i += 256) {
    float e = __expf(energy[i] - maxv);
    energy[i] = e;
    sum += e;
  }
  red[tid] = sum;
  __syncthreads();
  for (int s = 128; s > 0; s >>= 1) {
    if (tid < s) red[tid] += red[tid + s];
    __syncthreads();
  }
  float inv = 1.f / red[0];
  __syncthreads();
  float acc = 0.f;
  for (int tp = 0; tp < t; ++tp)
    acc += energy[tp] * (float)Hbf[((size_t)tp * B_ + b) * D_ + tid];
  ctx_out[tid] = (__bf16)(acc * inv);
}

// logits[b][t][:] = HC[t*B+b][:] @ fc_W + fc_b  (M=32768, N=512, K=512)
// A panel (16x512 bf16, 16KB) staged in LDS via TDM; 8 waves x 4 N-tiles.
__global__ void __launch_bounds__(256) k_logits(const __bf16* __restrict__ HC,
                                                const __bf16* __restrict__ fcWT,
                                                const float* __restrict__ fc_b,
                                                float* __restrict__ out) {
  __shared__ __bf16 a_s[16 * 2 * D_];  // 16 KB
  int tid = threadIdx.x;
  int lane = tid & 31, wave = tid >> 5;
  int m0 = blockIdx.x * 16;  // 2048 M-tiles
  const __bf16* Apanel = HC + (size_t)m0 * (2 * D_);
#ifdef HAVE_TDM
  if (wave == 0) tdm_load_panel(Apanel, a_s, 16, 2 * D_);
#else
  for (int i = tid * 8; i < 16 * 2 * D_; i += 256 * 8)
    *(v8bf*)(a_s + i) = *(const v8bf*)(Apanel + i);
#endif
  __syncthreads();
#pragma unroll
  for (int it = 0; it < 4; ++it) {
    int n0 = (wave * 4 + it) * 16;  // 32 N-tiles per block
    v8f acc = {};
#pragma unroll
    for (int k0 = 0; k0 < 2 * D_; k0 += 32) {
      v16bf a = load_a16(a_s, 2 * D_, 0, k0, lane);
      v16bf b = load_bT16(fcWT, 2 * D_, n0, k0, lane);
      acc = wmma_bf16(a, b, acc);
    }
    int n = n0 + (lane & 15);
    int mB = m0 + ((lane >> 4) << 3);
    float bias = fc_b[n];
#pragma unroll
    for (int v = 0; v < 8; ++v) {
      int m = mB + v;  // m = t*B + b
      int tt = m >> 6, bb = m & 63;
      out[((size_t)bb * T_ + tt) * V_ + n] = acc[v] + bias;
    }
  }
}

extern "C" void kernel_launch(void* const* d_in, const int* in_sizes, int n_in,
                              void* d_out, int out_size, void* d_ws, size_t ws_size,
                              hipStream_t stream) {
  (void)in_sizes; (void)n_in; (void)out_size; (void)ws_size;
  const int* x      = (const int*)d_in[0];
  const float* embed  = (const float*)d_in[1];
  const float* W_ih   = (const float*)d_in[2];
  const float* W_hh   = (const float*)d_in[3];
  const float* b_h    = (const float*)d_in[4];
  const float* W_attn = (const float*)d_in[5];
  const float* U_attn = (const float*)d_in[6];
  const float* v_attn = (const float*)d_in[7];
  const float* fc_W   = (const float*)d_in[8];
  const float* fc_b   = (const float*)d_in[9];
  float* out = (float*)d_out;

  char* ws = (char*)d_ws;
  size_t off = 0;
  auto take = [&](size_t bytes) {
    void* p = ws + off;
    off += (bytes + 255) & ~(size_t)255;
    return p;
  };
  float*  XP   = (float*)take((size_t)T_ * B_ * D_ * 4);
  float*  P    = (float*)take((size_t)T_ * B_ * D_ * 4);
  float*  Proj = (float*)take((size_t)B_ * D_ * 4);
  __bf16* Hbf  = (__bf16*)take((size_t)T_ * B_ * D_ * 2);
  __bf16* HC   = (__bf16*)take((size_t)T_ * B_ * 2 * D_ * 2);
  __bf16* Ebf  = (__bf16*)take((size_t)V_ * E_ * 2);
  __bf16* WihT = (__bf16*)take((size_t)D_ * E_ * 2);
  __bf16* WhhT = (__bf16*)take((size_t)D_ * D_ * 2);
  __bf16* UatT = (__bf16*)take((size_t)D_ * D_ * 2);
  __bf16* WatT = (__bf16*)take((size_t)D_ * D_ * 2);
  __bf16* fcWT = (__bf16*)take((size_t)V_ * 2 * D_ * 2);

  k_cvt_bf16<<<(V_ * E_ + 255) / 256, 256, 0, stream>>>(embed, Ebf, V_ * E_);
  k_transpose_bf16<<<(E_ * D_ + 255) / 256, 256, 0, stream>>>(W_ih, WihT, E_, D_);
  k_transpose_bf16<<<(D_ * D_ + 255) / 256, 256, 0, stream>>>(W_hh, WhhT, D_, D_);
  k_transpose_bf16<<<(D_ * D_ + 255) / 256, 256, 0, stream>>>(U_attn, UatT, D_, D_);
  k_transpose_bf16<<<(D_ * D_ + 255) / 256, 256, 0, stream>>>(W_attn, WatT, D_, D_);
  k_transpose_bf16<<<(2 * D_ * V_ + 255) / 256, 256, 0, stream>>>(fc_W, fcWT,
                                                                  2 * D_, V_);

  k_embed_proj<<<(B_ * T_) / 16, 256, 0, stream>>>(x, Ebf, WihT, b_h, XP);

  for (int t = 0; t < T_; ++t) {
    const __bf16* Hprev = (t > 0) ? (Hbf + (size_t)(t - 1) * B_ * D_) : Hbf;
    k_step<<<1, 256, 0, stream>>>(t, XP, Hprev, WhhT, UatT, WatT, b_h,
                                  Hbf + (size_t)t * B_ * D_,
                                  P + (size_t)t * B_ * D_, Proj,
                                  HC + (size_t)t * B_ * 2 * D_);
    k_attn<<<B_, 256, 0, stream>>>(t, P, Proj, Hbf, v_attn, HC);
  }

  k_logits<<<(B_ * T_) / 16, 256, 0, stream>>>(HC, fcWT, fc_b, out);
}
